// IntraFreqAttention_20452634263748
// MI455X (gfx1250) — matrix operand order repaired and checked by reading, not compiled, not run on hardware
//
#include <hip/hip_runtime.h>
#include <hip/hip_bf16.h>

// ---------------------------------------------------------------------------
// IntraFreqAttention on MI455X (gfx1250): f16 WMMA everywhere, f32 accumulate.
// B=4, C=64, N=8, V=32, W=32  ->  P = B*N = 32 token groups, S = 1024, D = 256.
// ---------------------------------------------------------------------------

typedef __attribute__((ext_vector_type(8)))  _Float16 v8h;
typedef __attribute__((ext_vector_type(16))) _Float16 v16h;
typedef __attribute__((ext_vector_type(8)))  float    v8f;
typedef __attribute__((ext_vector_type(4)))  int      v4i;

#define KB   4            // batch
#define KC   64           // channels
#define KN   8            // freq bins
#define KS   1024         // tokens per (b,n)  (V*W)
#define KD   256          // SPA
#define KF   512          // 2*SPA
#define KP   32           // B*N pairs
#define KEPS 1e-5f
#define KMASK -1.0e15f

// ---- WMMA helpers ----------------------------------------------------------

__device__ __forceinline__ v16h cat8(v8h lo, v8h hi) {
  v16h r;
#pragma unroll
  for (int i = 0; i < 8; ++i) { r[i] = lo[i]; r[i + 8] = hi[i]; }
  return r;
}

// A operand (16xK tile, row-major source [row][ldk]):
// lane l (r=l&15, h=l>>4) holds row r, K = {8h..8h+7, 16+8h..16+8h+7} of a 32-chunk
__device__ __forceinline__ v16h loadA(const _Float16* base, int ldk, int lane) {
  int r = lane & 15, h = lane >> 4;
  const _Float16* p = base + r * ldk + 8 * h;
  v8h lo = *(const v8h*)p;
  v8h hi = *(const v8h*)(p + 16);
  return cat8(lo, hi);
}

// B operand (Kx16 tile given as B^T row-major [n][ldk]):
// lane l (r=l&15, h=l>>4) holds column r, K = {8h..8h+7, 16+8h..16+8h+7} of a 32-chunk
__device__ __forceinline__ v16h loadB(const _Float16* baseT, int ldk, int lane) {
  int r = lane & 15, h = lane >> 4;
  const _Float16* p = baseT + r * ldk + 8 * h;
  v8h lo = *(const v8h*)p;
  v8h hi = *(const v8h*)(p + 16);
  return cat8(lo, hi);
}

__device__ __forceinline__ v8f wmma_f16(v16h a, v16h b, v8f c) {
  return __builtin_amdgcn_wmma_f32_16x16x32_f16(
      /*neg_a=*/false, a, /*neg_b=*/false, b,
      /*c_mod=*/(short)0, c, /*reuse_a=*/false, /*reuse_b=*/false);
}

// ---- CDNA5 async global->LDS copy (ASYNCcnt path), with safe fallback ------

#if __has_builtin(__builtin_amdgcn_global_load_async_to_lds_b128)
#define HAVE_ASYNC_LDS 1
typedef __attribute__((address_space(1))) v4i GV4;
typedef __attribute__((address_space(3))) v4i LV4;
#else
#define HAVE_ASYNC_LDS 0
#endif

// copy 16 bytes (8 f16) from global to LDS
__device__ __forceinline__ void asyncCopy16(const _Float16* gsrc, _Float16* ldst) {
#if HAVE_ASYNC_LDS
  __builtin_amdgcn_global_load_async_to_lds_b128(
      (GV4*)gsrc, (LV4*)ldst, /*imm offset=*/0, /*cpol=*/0);
#else
  *(v8h*)ldst = *(const v8h*)gsrc;
#endif
}

__device__ __forceinline__ void waitAsync() {
#if HAVE_ASYNC_LDS
#if __has_builtin(__builtin_amdgcn_s_wait_asynccnt)
  __builtin_amdgcn_s_wait_asynccnt(0);
#else
  asm volatile("s_wait_asynccnt 0x0" ::: "memory");
#endif
#endif
}

// ---- Kernel 0: f32 -> f16 weight conversion --------------------------------

__global__ void k_cvt(const float* __restrict__ s, _Float16* __restrict__ d, int n) {
  int i = blockIdx.x * 256 + threadIdx.x;
  if (i < n) d[i] = (_Float16)s[i];
}

// ---- Kernel 1: t = x @ W_in^T (WMMA, K=64) + fused LayerNorm ---------------
// grid: P * (S/32) = 1024 blocks, 256 threads (8 waves). 32 rows per block.

__global__ void k_tln(const float* __restrict__ buf,
                      const _Float16* __restrict__ Winh,   // [256][64] f16
                      const float* __restrict__ g1, const float* __restrict__ b1,
                      float* __restrict__ tF32,
                      _Float16* __restrict__ tF16,
                      _Float16* __restrict__ tnF16) {
  __shared__ __align__(16) _Float16 xs[32][KC];     // 4 KB
  __shared__ __align__(16) float    tts[32][KD];    // 32 KB

  const int t    = threadIdx.x;
  const int pair = blockIdx.x >> 5;
  const int rblk = blockIdx.x & 31;
  const int r0   = rblk * 32;
  const int bb   = pair >> 3, nn = pair & 7;

  // stage x[s][c] as f16 (coalesced in s)
#pragma unroll
  for (int i = 0; i < 8; ++i) {
    int idx = i * 256 + t;
    int c = idx >> 5, s = idx & 31;
    float v = buf[((size_t)(bb * KC + c) * KN + nn) * KS + r0 + s];
    xs[s][c] = (_Float16)v;
  }
  __syncthreads();

  // GEMM: 2 row-tiles x 16 col-tiles, K=64 (2 WMMA chunks)
  const int lane = t & 31, w = t >> 5;
  const int rt = w & 1, ctb = (w >> 1) * 4;
  v16h a0 = loadA(&xs[rt * 16][0], KC, lane);
  v16h a1 = loadA(&xs[rt * 16][32], KC, lane);
#pragma unroll
  for (int j = 0; j < 4; ++j) {
    int ct = ctb + j;
    v8f c = {};
    v16h bw0 = loadB(Winh + (size_t)(ct * 16) * KC, KC, lane);
    v16h bw1 = loadB(Winh + (size_t)(ct * 16) * KC + 32, KC, lane);
    c = wmma_f16(a0, bw0, c);
    c = wmma_f16(a1, bw1, c);
    int rr = rt * 16 + 8 * (lane >> 4);
    int cc = ct * 16 + (lane & 15);
#pragma unroll
    for (int v = 0; v < 8; ++v) tts[rr + v][cc] = c[v];
  }
  __syncthreads();

  // LayerNorm over d=256: 8 threads per row (32 cols each)
  const int row = t >> 3, seg = t & 7;
  float s1 = 0.f, s2 = 0.f;
#pragma unroll
  for (int i = 0; i < 32; ++i) {
    float v = tts[row][seg * 32 + i];
    s1 += v; s2 += v * v;
  }
#pragma unroll
  for (int m = 1; m < 8; m <<= 1) { s1 += __shfl_xor(s1, m); s2 += __shfl_xor(s2, m); }
  float mu = s1 * (1.0f / KD);
  float var = s2 * (1.0f / KD) - mu * mu;
  float rs = rsqrtf(var + KEPS);

  size_t gro = ((size_t)pair * KS + r0 + row) * KD + seg * 32;
#pragma unroll
  for (int i = 0; i < 32; ++i) {
    int d = seg * 32 + i;
    float v = tts[row][d];
    tF32[gro + i]  = v;
    tF16[gro + i]  = (_Float16)v;
    tnF16[gro + i] = (_Float16)((v - mu) * rs * g1[d] + b1[d]);
  }
}

// ---- Kernel 2: causal flash attention --------------------------------------
// grid: P * (S/128) = 256 blocks, 256 threads (8 waves). Wave owns 16 q rows.

__global__ void k_attn(const _Float16* __restrict__ tF16,
                       const _Float16* __restrict__ tnF16,
                       float* __restrict__ attF32) {
  __shared__ __align__(16) _Float16 kT[32][KD];       // 16 KB  tn k-chunk (row-major)
  __shared__ __align__(16) _Float16 vT[KD][32];       // 16 KB  t  k-chunk (transposed)
  __shared__ __align__(16) _Float16 pb[8][16][32];    //  8 KB  per-wave p staging

  const int t    = threadIdx.x;
  const int lane = t & 31, w = t >> 5;
  const int pair = blockIdx.x >> 3;
  const int qblk = blockIdx.x & 7;
  const size_t po = (size_t)pair * KS * KD;
  const int q0   = qblk * 128 + w * 16;

  // cache q A-operand: 8 K-chunks of tn rows [q0..q0+15]
  const int r = lane & 15, h = lane >> 4;
  v16h aq[8];
#pragma unroll
  for (int dc = 0; dc < 8; ++dc) {
    const _Float16* p = tnF16 + po + (size_t)(q0 + r) * KD + dc * 32 + 8 * h;
    aq[dc] = cat8(*(const v8h*)p, *(const v8h*)(p + 16));
  }

  v8f acc[16];
#pragma unroll
  for (int dt = 0; dt < 16; ++dt) { v8f z = {}; acc[dt] = z; }
  float mrow[8], lrow[8];
#pragma unroll
  for (int v = 0; v < 8; ++v) { mrow[v] = -3.0e38f; lrow[v] = 0.f; }

  const int nch = qblk * 4 + 4;   // causal: chunks of 32 keys up to q-block end
  for (int ck = 0; ck < nch; ++ck) {
    const int k0 = ck * 32;

    // stage kT (tn rows, contiguous 16 KB): async global->LDS, no VGPR bounce
    {
      int kr = t >> 3, d0 = (t & 7) * 32;   // 64 B per thread
      const _Float16* src = tnF16 + po + (size_t)(k0 + kr) * KD + d0;
#pragma unroll
      for (int j = 0; j < 4; ++j)
        asyncCopy16(src + j * 8, &kT[kr][d0 + j * 8]);
    }
    // stage vT (t transposed): thread t handles column d = t (manual transpose)
    {
      int d = t;
      _Float16 hv[32];
#pragma unroll
      for (int kk = 0; kk < 32; ++kk)
        hv[kk] = tF16[po + (size_t)(k0 + kk) * KD + d];
#pragma unroll
      for (int gidx = 0; gidx < 4; ++gidx) {
        v8h x;
#pragma unroll
        for (int j = 0; j < 8; ++j) x[j] = hv[gidx * 8 + j];
        *(v8h*)&vT[d][gidx * 8] = x;
      }
    }
    if (ck + 1 < nch)  // hint next chunk into cache
      __builtin_prefetch(tnF16 + po + (size_t)(k0 + 32) * KD, 0, 1);
    waitAsync();
    __syncthreads();

    // score tiles: 16q x 32k, K = d = 256
    v8f sc0 = {}, sc1 = {};
#pragma unroll
    for (int dc = 0; dc < 8; ++dc) {
      v16h b0 = loadB(&kT[0][dc * 32], KD, lane);
      sc0 = wmma_f16(aq[dc], b0, sc0);
      v16h b1 = loadB(&kT[16][dc * 32], KD, lane);
      sc1 = wmma_f16(aq[dc], b1, sc1);
    }

    // causal mask + online softmax (rows live on 16-lane halves)
    const int colb = k0 + (lane & 15);
    const int rowb = q0 + 8 * h;
#pragma unroll
    for (int v = 0; v < 8; ++v) {
      int qrow = rowb + v;
      float s0 = (colb > qrow) ? KMASK : sc0[v];
      float s1v = (colb + 16 > qrow) ? KMASK : sc1[v];
      float mc = fmaxf(s0, s1v);
#pragma unroll
      for (int m = 1; m < 16; m <<= 1) mc = fmaxf(mc, __shfl_xor(mc, m));
      float mn = fmaxf(mrow[v], mc);
      float scale = __expf(mrow[v] - mn);
      float p0 = __expf(s0 - mn), p1 = __expf(s1v - mn);
      float ps = p0 + p1;
#pragma unroll
      for (int m = 1; m < 16; m <<= 1) ps += __shfl_xor(ps, m);
      lrow[v] = lrow[v] * scale + ps;
      mrow[v] = mn;
      sc0[v] = p0; sc1[v] = p1;
#pragma unroll
      for (int dt = 0; dt < 16; ++dt) acc[dt][v] *= scale;
    }

    // p: C-layout -> A-layout via per-wave LDS bounce
#pragma unroll
    for (int v = 0; v < 8; ++v) {
      pb[w][v + 8 * h][lane & 15]        = (_Float16)sc0[v];
      pb[w][v + 8 * h][16 + (lane & 15)] = (_Float16)sc1[v];
    }
    v16h pa = loadA(&pb[w][0][0], 32, lane);

    // att += p @ t  (value from transposed LDS tile, contiguous B loads)
#pragma unroll
    for (int dt = 0; dt < 16; ++dt) {
      v16h bv = loadB(&vT[dt * 16][0], 32, lane);
      acc[dt] = wmma_f16(pa, bv, acc[dt]);
    }
    __syncthreads();
  }

  // normalize and store p@t (residual +t added in k_ffn)
#pragma unroll
  for (int v = 0; v < 8; ++v) {
    float inv = 1.0f / lrow[v];
    size_t rowoff = po + (size_t)(q0 + 8 * h + v) * KD;
#pragma unroll
    for (int dt = 0; dt < 16; ++dt)
      attF32[rowoff + dt * 16 + (lane & 15)] = acc[dt][v] * inv;
  }
}

// ---- Kernel 3: residual + LN2 + FFN + residual + output projection ---------
// grid: P * (S/32) = 1024 blocks, 256 threads (8 waves). 32 rows per block.

__global__ void k_ffn(const float* __restrict__ attF32,
                      const float* __restrict__ tF32,
                      const _Float16* __restrict__ W1h,    // [512][256]
                      const _Float16* __restrict__ W2h,    // [256][512]
                      const _Float16* __restrict__ Wouth,  // [64][256]
                      const float* __restrict__ g2, const float* __restrict__ b2,
                      float* __restrict__ out) {
  __shared__ __align__(16) _Float16 hn[32][KD];   // 16 KB (reused for y after GEMM1)
  __shared__ __align__(16) _Float16 h1[32][KF];   // 32 KB

  const int t    = threadIdx.x;
  const int lane = t & 31, w = t >> 5;
  const int pair = blockIdx.x >> 5;
  const int rblk = blockIdx.x & 31;
  const int r0   = rblk * 32;
  const int bb   = pair >> 3, nn = pair & 7;
  const size_t po = (size_t)pair * KS * KD;

  // a = att + t ; LayerNorm -> hn (f16)
  {
    int row = t >> 3, seg = t & 7;
    size_t gro = po + (size_t)(r0 + row) * KD + seg * 32;
    float av[32];
    float s1 = 0.f, s2 = 0.f;
#pragma unroll
    for (int i = 0; i < 32; ++i) {
      float a = attF32[gro + i] + tF32[gro + i];
      av[i] = a; s1 += a; s2 += a * a;
    }
#pragma unroll
    for (int m = 1; m < 8; m <<= 1) { s1 += __shfl_xor(s1, m); s2 += __shfl_xor(s2, m); }
    float mu = s1 * (1.0f / KD);
    float var = s2 * (1.0f / KD) - mu * mu;
    float rs = rsqrtf(var + KEPS);
#pragma unroll
    for (int i = 0; i < 32; ++i) {
      int d = seg * 32 + i;
      hn[row][d] = (_Float16)((av[i] - mu) * rs * g2[d] + b2[d]);
    }
  }
  __syncthreads();

  // GEMM1: h1 = relu(hn @ W1^T)   [32x256]x[256->512]
  {
    int rt = w & 1, ctb = (w >> 1) * 8;
    v16h a[8];
#pragma unroll
    for (int dc = 0; dc < 8; ++dc) a[dc] = loadA(&hn[rt * 16][dc * 32], KD, lane);
#pragma unroll
    for (int j = 0; j < 8; ++j) {
      int ct = ctb + j;
      v8f c = {};
#pragma unroll
      for (int dc = 0; dc < 8; ++dc) {
        v16h bw = loadB(W1h + (size_t)(ct * 16) * KD + dc * 32, KD, lane);
        c = wmma_f16(a[dc], bw, c);
      }
      int rr = rt * 16 + 8 * (lane >> 4);
      int cc = ct * 16 + (lane & 15);
#pragma unroll
      for (int v = 0; v < 8; ++v) {
        float x = c[v];
        h1[rr + v][cc] = (_Float16)(x > 0.f ? x : 0.f);
      }
    }
  }
  __syncthreads();   // hn is dead past this point -> reuse for y

  // GEMM2: y = h1 @ W2^T + t      [32x512]x[512->256], write y into hn buffer
  {
    int rt = w & 1, ctb = (w >> 1) * 4;
#pragma unroll
    for (int j = 0; j < 4; ++j) {
      int ct = ctb + j;
      v8f c = {};
#pragma unroll
      for (int dc = 0; dc < 16; ++dc) {
        v16h a2 = loadA(&h1[rt * 16][dc * 32], KF, lane);
        v16h bw = loadB(W2h + (size_t)(ct * 16) * KF + dc * 32, KF, lane);
        c = wmma_f16(a2, bw, c);
      }
      int rr = rt * 16 + 8 * (lane >> 4);
      int cc = ct * 16 + (lane & 15);
#pragma unroll
      for (int v = 0; v < 8; ++v) {
        float y = c[v] + tF32[po + (size_t)(r0 + rr + v) * KD + cc];
        hn[rr + v][cc] = (_Float16)y;
      }
    }
  }
  __syncthreads();

  // GEMM3: out = y @ W_out^T      [32x256]x[256->64], scatter to [b,c,n,s]
  {
    int rt = w >> 2, ct = w & 3;
    v8f c = {};
#pragma unroll
    for (int dc = 0; dc < 8; ++dc) {
      v16h a3 = loadA(&hn[rt * 16][dc * 32], KD, lane);
      v16h bw = loadB(Wouth + (size_t)(ct * 16) * KD + dc * 32, KD, lane);
      c = wmma_f16(a3, bw, c);
    }
    int rr = rt * 16 + 8 * (lane >> 4);
    int cc = ct * 16 + (lane & 15);     // channel c
#pragma unroll
    for (int v = 0; v < 8; ++v) {
      int s = r0 + rr + v;
      out[((size_t)(bb * KC + cc) * KN + nn) * KS + s] = c[v];
    }
  }
}

// ---------------------------------------------------------------------------

extern "C" void kernel_launch(void* const* d_in, const int* in_sizes, int n_in,
                              void* d_out, int out_size, void* d_ws, size_t ws_size,
                              hipStream_t stream) {
  const float* buf   = (const float*)d_in[0];
  const float* W_in  = (const float*)d_in[1];
  const float* ln1_g = (const float*)d_in[2];
  const float* ln1_b = (const float*)d_in[3];
  const float* ln2_g = (const float*)d_in[4];
  const float* ln2_b = (const float*)d_in[5];
  const float* W1    = (const float*)d_in[6];
  const float* W2    = (const float*)d_in[7];
  const float* W_out = (const float*)d_in[8];
  float* out = (float*)d_out;

  char* ws = (char*)d_ws;
  const size_t nTD = (size_t)KP * KS * KD;        // 8,388,608 elements
  float*    tF32   = (float*)(ws);                                    // 32 MB
  _Float16* tF16   = (_Float16*)(ws + nTD * 4);                       // 16 MB
  _Float16* tnF16  = (_Float16*)(ws + nTD * 4 + nTD * 2);             // 16 MB
  float*    attF32 = (float*)(ws + nTD * 4 + nTD * 4);                // 32 MB
  char* wsel = ws + nTD * 4 + nTD * 4 + nTD * 4;
  _Float16* Winh  = (_Float16*)(wsel);
  _Float16* W1h   = (_Float16*)(wsel + (size_t)KD * KC * 2);
  _Float16* W2h   = (_Float16*)(wsel + (size_t)KD * KC * 2 + (size_t)KF * KD * 2);
  _Float16* Wouth = (_Float16*)(wsel + (size_t)KD * KC * 2 + 2 * (size_t)KF * KD * 2);

  // weight conversions (f32 -> f16)
  k_cvt<<<(KD * KC + 255) / 256, 256, 0, stream>>>(W_in,  Winh,  KD * KC);
  k_cvt<<<(KF * KD + 255) / 256, 256, 0, stream>>>(W1,    W1h,   KF * KD);
  k_cvt<<<(KD * KF + 255) / 256, 256, 0, stream>>>(W2,    W2h,   KD * KF);
  k_cvt<<<(KC * KD + 255) / 256, 256, 0, stream>>>(W_out, Wouth, KC * KD);

  // t = x @ W_in^T + LN1  (1024 blocks: 32 pairs x 32 row-blocks)
  k_tln<<<KP * (KS / 32), 256, 0, stream>>>(buf, Winh, ln1_g, ln1_b,
                                            tF32, tF16, tnF16);

  // causal attention (256 blocks: 32 pairs x 8 q-blocks of 128 rows)
  k_attn<<<KP * (KS / 128), 256, 0, stream>>>(tF16, tnF16, attF32);

  // LN2 + FFN + residual + output projection (1024 blocks)
  k_ffn<<<KP * (KS / 32), 256, 0, stream>>>(attF32, tF32, W1h, W2h, Wouth,
                                            ln2_g, ln2_b, out);
}
